// SNN_60438779789796
// MI455X (gfx1250) — matrix operand order repaired and checked by reading, not compiled, not run on hardware
//
#include <hip/hip_runtime.h>
#include <hip/hip_bf16.h>

// ---------------- types for WMMA ----------------
typedef __attribute__((ext_vector_type(16))) _Float16 v16h;
typedef __attribute__((ext_vector_type(8)))  _Float16 v8h;
typedef __attribute__((ext_vector_type(8)))  float    v8f;

#define T_STEPS 200
#define BATCH   256
#define NI      784
#define NH      4096
#define NO      10
#define KP      800          // NI padded to multiple of 32
#define BETA    0.99f
#define THR     1.0f

// ws layout (byte offsets, 16B aligned)
#define WS_XH    0ull                                  // [T*B, KP] f16 : 81,920,000 B
#define WS_W1H   81920000ull                           // [NH, KP] f16  :  6,553,600 B
#define WS_W2T   88473600ull                           // [NH, NO] f32  :    163,840 B
#define WS_CUR2  88637440ull                           // [T*B, NO] f32 :  2,048,000 B

// ---------------- helpers ----------------
__device__ inline v16h ld16(const _Float16* p0, const _Float16* p1) {
  v8h lo = *(const v8h*)p0;
  v8h hi = *(const v8h*)p1;
  v16h r;
#pragma unroll
  for (int i = 0; i < 8; ++i) { r[i] = lo[i]; r[i + 8] = hi[i]; }
  return r;
}

// ---------------- kernel 1/2: pad + convert f32 -> f16 ----------------
__global__ void cvt_pad_f16(const float* __restrict__ src, _Float16* __restrict__ dst,
                            int rows, int K, int Kpad) {
  long long i = (long long)blockIdx.x * blockDim.x + threadIdx.x;
  long long n = (long long)rows * Kpad;
  if (i >= n) return;
  int r = (int)(i / Kpad);
  int k = (int)(i % Kpad);
  float v = (k < K) ? src[(long long)r * K + k] : 0.0f;
  dst[i] = (_Float16)v;
}

// ---------------- kernel: transpose W2 [NO,NH] -> [NH,NO] ----------------
__global__ void build_w2t(const float* __restrict__ W2, float* __restrict__ W2T) {
  int i = blockIdx.x * blockDim.x + threadIdx.x;
  if (i >= NO * NH) return;
  int o = i / NH, k = i % NH;
  W2T[k * NO + o] = W2[i];
}

// ---------------- fused layer-1: WMMA GEMM + LIF recurrence ----------------
// grid: (NH/64, BATCH/128), block: 256 threads (8 waves: 4 along M x 2 along N)
// each wave: 32x32 tile = 2x2 fragments of 16x16, mem1 state held in VGPRs across t.
__global__ __launch_bounds__(256, 1) void snn_layer1(
    const _Float16* __restrict__ xh,   // [T*B, KP]
    const _Float16* __restrict__ w1h,  // [NH, KP]
    const float*    __restrict__ b1,   // [NH]
    float*          __restrict__ spk1) // [T, B, NH]
{
  const int lane    = threadIdx.x & 31;
  const int wave    = threadIdx.x >> 5;
  const int wm      = wave & 3;   // wave row (M)
  const int wn      = wave >> 2;  // wave col (N)
  const int lane15  = lane & 15;
  const int halfSel = lane >> 4;  // 0: lanes 0-15, 1: lanes 16-31

  const int hBase = blockIdx.x * 64 + wn * 32;   // hidden columns
  const int bBase = blockIdx.y * 128 + wm * 32;  // batch rows

  // per-nTile bias (each lane's 8 C elements share one column)
  float bias[2];
  bias[0] = b1[hBase + 0  + lane15];
  bias[1] = b1[hBase + 16 + lane15];

  // B-operand row pointers (W1 row h supplies column n=lane15; 16 contiguous K per lane)
  const _Float16* w1r[2];
  w1r[0] = w1h + (size_t)(hBase + 0  + lane15) * KP + halfSel * 16;
  w1r[1] = w1h + (size_t)(hBase + 16 + lane15) * KP + halfSel * 16;

  // A-operand row indices (x row b supplies A row m=lane15)
  const int arow[2] = { bBase + lane15, bBase + 16 + lane15 };

  // persistent membrane state: [mTile][nTile] of 16x16 f32 fragments
  v8f mem[2][2];
#pragma unroll
  for (int mt = 0; mt < 2; ++mt)
#pragma unroll
    for (int nt = 0; nt < 2; ++nt)
#pragma unroll
      for (int i = 0; i < 8; ++i) mem[mt][nt][i] = 0.0f;

#pragma unroll 1
  for (int t = 0; t < T_STEPS; ++t) {
    const _Float16* xr[2];
    xr[0] = xh + ((size_t)t * BATCH + arow[0]) * KP + halfSel * 16;
    xr[1] = xh + ((size_t)t * BATCH + arow[1]) * KP + halfSel * 16;

    // acc = beta*mem_old - reset(mem_old)*THR + b1
    v8f acc[2][2];
#pragma unroll
    for (int mt = 0; mt < 2; ++mt)
#pragma unroll
      for (int nt = 0; nt < 2; ++nt)
#pragma unroll
        for (int i = 0; i < 8; ++i) {
          float m = mem[mt][nt][i];
          acc[mt][nt][i] = BETA * m - ((m > THR) ? THR : 0.0f) + bias[nt];
        }

    // GEMM over K (25 steps of K=32), accumulate into acc
#pragma unroll 1
    for (int kk = 0; kk < KP; kk += 32) {
      v16h a0 = ld16(xr[0] + kk, xr[0] + kk + 8);
      v16h a1 = ld16(xr[1] + kk, xr[1] + kk + 8);
      v16h bb0 = ld16(w1r[0] + kk, w1r[0] + kk + 8);
      v16h bb1 = ld16(w1r[1] + kk, w1r[1] + kk + 8);

      acc[0][0] = __builtin_amdgcn_wmma_f32_16x16x32_f16(false, a0, false, bb0,
                                                         (short)0, acc[0][0], false, false);
      acc[0][1] = __builtin_amdgcn_wmma_f32_16x16x32_f16(false, a0, false, bb1,
                                                         (short)0, acc[0][1], false, false);
      acc[1][0] = __builtin_amdgcn_wmma_f32_16x16x32_f16(false, a1, false, bb0,
                                                         (short)0, acc[1][0], false, false);
      acc[1][1] = __builtin_amdgcn_wmma_f32_16x16x32_f16(false, a1, false, bb1,
                                                         (short)0, acc[1][1], false, false);
    }

    // commit state and emit spikes
    const size_t outBase = (size_t)t * BATCH * NH;
#pragma unroll
    for (int mt = 0; mt < 2; ++mt)
#pragma unroll
      for (int nt = 0; nt < 2; ++nt) {
        mem[mt][nt] = acc[mt][nt];
        const int h = hBase + nt * 16 + lane15;
#pragma unroll
        for (int j = 0; j < 8; ++j) {
          const int b = bBase + mt * 16 + halfSel * 8 + j;
          spk1[outBase + (size_t)b * NH + h] = (mem[mt][nt][j] > THR) ? 1.0f : 0.0f;
        }
      }
  }
}

// ---------------- layer-2 current: cur2 = spk1 @ W2^T + b2 ----------------
// one wave per (t,b) row; memory-bound on spk1 (839 MB); W2T stays in L0/L2.
__global__ __launch_bounds__(256) void snn_layer2_cur(
    const float* __restrict__ spk1,  // [T*B, NH]
    const float* __restrict__ w2t,   // [NH, NO]
    const float* __restrict__ b2,    // [NO]
    float*       __restrict__ cur2)  // [T*B, NO]
{
  const int row  = blockIdx.x * 8 + (threadIdx.x >> 5);
  const int lane = threadIdx.x & 31;
  const float* sr = spk1 + (size_t)row * NH;

  float acc[NO];
#pragma unroll
  for (int o = 0; o < NO; ++o) acc[o] = 0.0f;

  for (int k = lane * 4; k < NH; k += 128) {
    float4 s = *(const float4*)(sr + k);
    float sv[4] = { s.x, s.y, s.z, s.w };
#pragma unroll
    for (int e = 0; e < 4; ++e) {
      const float sc = sv[e];
      const float* wrow = w2t + (size_t)(k + e) * NO;
#pragma unroll
      for (int o = 0; o < NO; ++o) acc[o] = fmaf(sc, wrow[o], acc[o]);
    }
  }
#pragma unroll
  for (int off = 16; off > 0; off >>= 1)
#pragma unroll
    for (int o = 0; o < NO; ++o) acc[o] += __shfl_down(acc[o], off);

  if (lane == 0) {
#pragma unroll
    for (int o = 0; o < NO; ++o) cur2[(size_t)row * NO + o] = acc[o] + b2[o];
  }
}

// ---------------- layer-2 membrane recurrence ----------------
__global__ void snn_mem2(const float* __restrict__ cur2,
                         float* __restrict__ spk2,    // [T, B*NO]
                         float* __restrict__ mem2rec) // [T, B*NO]
{
  const int idx = blockIdx.x * blockDim.x + threadIdx.x;
  const int n = BATCH * NO;
  if (idx >= n) return;
  float m = 0.0f;
#pragma unroll 1
  for (int t = 0; t < T_STEPS; ++t) {
    float c = cur2[(size_t)t * n + idx];
    float reset = (m > THR) ? THR : 0.0f;
    m = BETA * m + c - reset;
    spk2[(size_t)t * n + idx]    = (m > THR) ? 1.0f : 0.0f;
    mem2rec[(size_t)t * n + idx] = m;
  }
}

// ---------------- launcher ----------------
extern "C" void kernel_launch(void* const* d_in, const int* in_sizes, int n_in,
                              void* d_out, int out_size, void* d_ws, size_t ws_size,
                              hipStream_t stream) {
  (void)in_sizes; (void)n_in; (void)out_size; (void)ws_size;

  const float* x  = (const float*)d_in[0];  // [T, B, NI]
  const float* W1 = (const float*)d_in[1];  // [NH, NI]
  const float* b1 = (const float*)d_in[2];  // [NH]
  const float* W2 = (const float*)d_in[3];  // [NO, NH]
  const float* b2 = (const float*)d_in[4];  // [NO]

  float* out  = (float*)d_out;
  float* spk1 = out;                                            // T*B*NH
  float* spk2 = out + (size_t)T_STEPS * BATCH * NH;             // T*B*NO
  float* mem2 = spk2 + (size_t)T_STEPS * BATCH * NO;            // T*B*NO

  char* ws = (char*)d_ws;
  _Float16* xh   = (_Float16*)(ws + WS_XH);
  _Float16* w1h  = (_Float16*)(ws + WS_W1H);
  float*    w2t  = (float*)(ws + WS_W2T);
  float*    cur2 = (float*)(ws + WS_CUR2);

  // 1) convert/pad x : rows = T*B = 51200, K 784 -> 800
  {
    long long n = (long long)T_STEPS * BATCH * KP;  // 40,960,000
    int blocks = (int)((n + 255) / 256);
    cvt_pad_f16<<<blocks, 256, 0, stream>>>(x, xh, T_STEPS * BATCH, NI, KP);
  }
  // 2) convert/pad W1 : rows = NH
  {
    long long n = (long long)NH * KP;               // 3,276,800
    int blocks = (int)((n + 255) / 256);
    cvt_pad_f16<<<blocks, 256, 0, stream>>>(W1, w1h, NH, NI, KP);
  }
  // 3) transpose W2
  {
    int n = NO * NH;
    build_w2t<<<(n + 255) / 256, 256, 0, stream>>>(W2, w2t);
  }
  // 4) fused layer-1 (WMMA + LIF)
  {
    dim3 grid(NH / 64, BATCH / 128);  // (64, 2)
    snn_layer1<<<grid, 256, 0, stream>>>(xh, w1h, b1, spk1);
  }
  // 5) layer-2 currents
  {
    int rows = T_STEPS * BATCH;       // 51200, 8 rows per block
    snn_layer2_cur<<<rows / 8, 256, 0, stream>>>(spk1, w2t, b2, cur2);
  }
  // 6) layer-2 recurrence + outputs
  {
    int n = BATCH * NO;               // 2560
    snn_mem2<<<(n + 255) / 256, 256, 0, stream>>>(cur2, spk2, mem2);
  }
}
